// RaftModule_7980049236690
// MI455X (gfx1250) — matrix operand order repaired and checked by reading, not compiled, not run on hardware
//
#include <hip/hip_runtime.h>
#include <hip/hip_bf16.h>
#include <stdint.h>

typedef _Float16 half_t;
typedef __attribute__((ext_vector_type(16))) _Float16 v16h;
typedef __attribute__((ext_vector_type(8)))  float    v8f;

#define BM 64
#define BN 64     // conv kernel N tile
#define BNG 128   // correlation GEMM N tile
#define BK 32

// Async global->LDS staging of one 16-byte chunk (CDNA5 GLOBAL_LOAD_ASYNC_TO_LDS_B128,
// tracked by ASYNCcnt). In CU-mode flat addressing the low 32 bits of a flat LDS
// pointer are the LDS byte offset, so the uint cast gives a valid VDST address.
__device__ __forceinline__ void async_b128(void* lds, const void* g) {
  uint32_t l = (uint32_t)(uintptr_t)lds;
  asm volatile("global_load_async_to_lds_b128 %0, %1, off" :: "v"(l), "v"(g) : "memory");
}
__device__ __forceinline__ void async_wait0() {
  asm volatile("s_wait_asynccnt 0" ::: "memory");
}

// ---------------------------------------------------------------------------
// Implicit-GEMM convolution with double-buffered staging:
//   out[b,co,oy,ox] = act( sum_k A(m,k)*W(k,co) + bias )
//   A(m,k) gathered on the fly from NCHW fp32 input, converted to f16 in LDS
//   (VALU path: the gather converts fp32->f16). Weight tile staged with async
//   b128 (fallback to guarded scalar staging at edges / unaligned N).
//   Software pipeline: issue async loads for tile t+1, compute WMMAs on tile t,
//   then s_wait_asynccnt + barrier -> global latency hides behind matrix ops.
//   Block: 256 thr = 8 waves (4x2), tile 64x64; each wave owns 16x32 output
//   (2 accumulators sharing one A fragment) -> 2 WMMA per K-step.
// ---------------------------------------------------------------------------
__global__ __launch_bounds__(256) void k_conv_wmma(
    const float* __restrict__ in, const half_t* __restrict__ wt,
    const float* __restrict__ bias, float* __restrict__ out,
    int Bn, int Cin, int H, int W, int Cout, int KH, int KW,
    int stride, int padh, int padw, int OH, int OW, int act)
{
  const int M = Bn * OH * OW, N = Cout, K = Cin * KH * KW;
  __shared__ __align__(16) half_t As[2][BM][BK + 8];   // row stride 80B
  __shared__ __align__(16) half_t Bs[2][BK][BN + 8];   // row stride 144B
  const int tid = threadIdx.x;
  const int lane = tid & 31, wid = tid >> 5;
  const int wm = wid & 3, wn = wid >> 2;               // 4 x 2 wave grid
  const int m0 = blockIdx.x * BM, n0 = blockIdx.y * BN;
  v8f acc0 = {}, acc1 = {};

  auto stageB = [&](int k0, int buf) {
    int r = tid >> 3, cc = (tid & 7) * 8;              // 32 rows x 64 cols, 16B/lane
    int k = k0 + r, n = n0 + cc;
    bool full = (k < K) && (n + 8 <= N) && ((N & 7) == 0);
    if (full) {
      async_b128(&Bs[buf][r][cc], wt + (size_t)k * N + n);
    } else {
#pragma unroll
      for (int e = 0; e < 8; ++e) {
        half_t v = (half_t)0.f;
        if (k < K && n + e < N) v = wt[(size_t)k * N + n + e];
        Bs[buf][r][cc + e] = v;
      }
    }
  };
  auto stageA = [&](int k0, int buf) {
    for (int i = tid; i < BM * BK; i += 256) {
      int r = i / BK, c = i % BK;
      int m = m0 + r, k = k0 + c;
      float v = 0.f;
      if (m < M && k < K) {
        int b   = m / (OH * OW); int rem = m % (OH * OW);
        int oy  = rem / OW,  ox = rem % OW;
        int ci  = k / (KH * KW); int kr = k % (KH * KW);
        int ky  = kr / KW,  kx = kr % KW;
        int iy  = oy * stride - padh + ky;
        int ix  = ox * stride - padw + kx;
        if (iy >= 0 && iy < H && ix >= 0 && ix < W)
          v = in[(((size_t)b * Cin + ci) * H + iy) * W + ix];
      }
      As[buf][r][c] = (half_t)v;
    }
  };

  const int ntiles = (K + BK - 1) / BK;
  stageB(0, 0);
  stageA(0, 0);
  async_wait0();
  __syncthreads();

  for (int t = 0; t < ntiles; ++t) {
    int cur = t & 1, nxt = cur ^ 1;
    if (t + 1 < ntiles) {              // prefetch next tile (no wait yet)
      stageB((t + 1) * BK, nxt);
      stageA((t + 1) * BK, nxt);
    }
    // fragments: A and B both in contiguous k order, k = 16*(lane/16) + e
    v16h af, bf0, bf1;
    {
      int ml = wm * 16 + (lane & 15);
      int kb = (lane < 16) ? 0 : 16;
#pragma unroll
      for (int e = 0; e < 8; ++e) {
        af[e]     = As[cur][ml][kb + e];
        af[e + 8] = As[cur][ml][kb + 8 + e];
      }
      int nl  = wn * 32 + (lane & 15);
      int kb2 = (lane < 16) ? 0 : 16;
#pragma unroll
      for (int e = 0; e < 16; ++e) {
        bf0[e] = Bs[cur][kb2 + e][nl];
        bf1[e] = Bs[cur][kb2 + e][nl + 16];
      }
    }
    acc0 = __builtin_amdgcn_wmma_f32_16x16x32_f16(
        false, af, false, bf0, (short)0, acc0, false, false);
    acc1 = __builtin_amdgcn_wmma_f32_16x16x32_f16(
        false, af, false, bf1, (short)0, acc1, false, false);
    async_wait0();
    __syncthreads();
  }

  // epilogue: bias + optional relu, scatter back to NCHW
  int col0  = n0 + wn * 32 + (lane & 15);
  int rbase = m0 + wm * 16 + ((lane < 16) ? 0 : 8);
#pragma unroll
  for (int h2 = 0; h2 < 2; ++h2) {
    int col = col0 + h2 * 16;
    if (col >= N) continue;
    float bv = bias ? bias[col] : 0.f;
#pragma unroll
    for (int r = 0; r < 8; ++r) {
      int m = rbase + r;
      if (m < M) {
        float v = (h2 ? acc1[r] : acc0[r]) + bv;
        if (act == 1) v = fmaxf(v, 0.f);
        int b = m / (OH * OW); int rem = m % (OH * OW);
        int oy = rem / OW, ox = rem % OW;
        out[(((size_t)b * Cout + col) * OH + oy) * OW + ox] = v;
      }
    }
  }
}

// ---------------------------------------------------------------------------
// Correlation GEMM: C[MxN] = alpha * A[MxK](f16,rm) * B[KxN](f16,rm)
// (M=N=3072, K=256.) Fully async double-buffered staging; each wave owns a
// 16x64 output (4 accumulators sharing one A fragment) -> 4 WMMA per K-step.
// Block tile 64x128, 8 waves (4x2).
// ---------------------------------------------------------------------------
__global__ __launch_bounds__(256) void k_gemm_wmma(
    const half_t* __restrict__ A, const half_t* __restrict__ Bm,
    float* __restrict__ C, int M, int N, int K, float alpha)
{
  __shared__ __align__(16) half_t As[2][BM][BK + 8];    // row stride 80B
  __shared__ __align__(16) half_t Bs[2][BK][BNG + 8];   // row stride 272B
  const int tid = threadIdx.x;
  const int lane = tid & 31, wid = tid >> 5;
  const int wm = wid & 3, wn = wid >> 2;
  const int m0 = blockIdx.x * BM, n0 = blockIdx.y * BNG;
  v8f acc[4] = {};

  auto stageA = [&](int k0, int buf) {
    int r = tid >> 2, cc = (tid & 3) * 8;               // 64 rows x 32 cols
    int m = m0 + r, k = k0 + cc;
    bool full = (m < M) && (k + 8 <= K) && ((K & 7) == 0);
    if (full) {
      async_b128(&As[buf][r][cc], A + (size_t)m * K + k);
    } else {
#pragma unroll
      for (int e = 0; e < 8; ++e)
        As[buf][r][cc + e] = (m < M && k + e < K) ? A[(size_t)m * K + k + e]
                                                  : (half_t)0.f;
    }
  };
  auto stageB = [&](int k0, int buf) {
    int r = tid >> 3;                                   // 32 rows x 128 cols
    int k = k0 + r;
#pragma unroll
    for (int j = 0; j < 2; ++j) {                       // 2 x 16B chunks / lane
      int cc = (tid & 7) * 16 + j * 8;
      int n = n0 + cc;
      bool full = (k < K) && (n + 8 <= N) && ((N & 7) == 0);
      if (full) {
        async_b128(&Bs[buf][r][cc], Bm + (size_t)k * N + n);
      } else {
#pragma unroll
        for (int e = 0; e < 8; ++e)
          Bs[buf][r][cc + e] = (k < K && n + e < N) ? Bm[(size_t)k * N + n + e]
                                                    : (half_t)0.f;
      }
    }
  };

  const int ntiles = (K + BK - 1) / BK;
  stageA(0, 0);
  stageB(0, 0);
  async_wait0();
  __syncthreads();

  for (int t = 0; t < ntiles; ++t) {
    int cur = t & 1, nxt = cur ^ 1;
    if (t + 1 < ntiles) {
      stageA((t + 1) * BK, nxt);
      stageB((t + 1) * BK, nxt);
    }
    v16h af;
    {
      int ml = wm * 16 + (lane & 15);
      int kb = (lane < 16) ? 0 : 16;
#pragma unroll
      for (int e = 0; e < 8; ++e) {
        af[e]     = As[cur][ml][kb + e];
        af[e + 8] = As[cur][ml][kb + 8 + e];
      }
    }
    int nl0 = wn * 64 + (lane & 15);
    int kb2 = (lane < 16) ? 0 : 16;
#pragma unroll
    for (int j = 0; j < 4; ++j) {
      v16h bf;
#pragma unroll
      for (int e = 0; e < 16; ++e) bf[e] = Bs[cur][kb2 + e][nl0 + j * 16];
      acc[j] = __builtin_amdgcn_wmma_f32_16x16x32_f16(
          false, af, false, bf, (short)0, acc[j], false, false);
    }
    async_wait0();
    __syncthreads();
  }

  int rbase = m0 + wm * 16 + ((lane < 16) ? 0 : 8);
#pragma unroll
  for (int j = 0; j < 4; ++j) {
    int col = n0 + wn * 64 + (lane & 15) + j * 16;
    if (col >= N) continue;
#pragma unroll
    for (int r = 0; r < 8; ++r) {
      int m = rbase + r;
      if (m < M) C[(size_t)m * N + col] = alpha * acc[j][r];
    }
  }
}

// ---------------------------------------------------------------------------
// Small utility kernels
// ---------------------------------------------------------------------------
__global__ void k_w2half(const float* __restrict__ w, half_t* __restrict__ wt,
                         int K, int N) {
  int idx = blockIdx.x * blockDim.x + threadIdx.x;
  if (idx < K * N) {
    int k = idx / N, n = idx % N;
    wt[idx] = (half_t)w[(size_t)n * K + k];   // OIHW -> [K,N]
  }
}

__global__ void k_f2h(const float* __restrict__ x, half_t* __restrict__ y,
                      long long n) {
  long long i = (long long)blockIdx.x * blockDim.x + threadIdx.x;
  if (i < n) y[i] = (half_t)x[i];
}

// fmap [B,D,P] -> A [B,P,D] f16
__global__ void k_t_f2h(const float* __restrict__ x, half_t* __restrict__ y,
                        int Bn, int D, int P) {
  long long i = (long long)blockIdx.x * blockDim.x + threadIdx.x;
  long long tot = (long long)Bn * D * P;
  if (i < tot) {
    int p = i % P; int d = (i / P) % D; int b = i / ((long long)D * P);
    y[((size_t)b * P + p) * D + d] = (half_t)x[i];
  }
}

__global__ void k_instnorm(const float* __restrict__ x, float* __restrict__ y,
                           int HW, int relu) {
  const float* px = x + (size_t)blockIdx.x * HW;
  float*       py = y + (size_t)blockIdx.x * HW;
  __shared__ float s1[256], s2[256];
  float a = 0.f, q = 0.f;
  for (int i = threadIdx.x; i < HW; i += 256) { float v = px[i]; a += v; q += v * v; }
  s1[threadIdx.x] = a; s2[threadIdx.x] = q; __syncthreads();
  for (int s = 128; s > 0; s >>= 1) {
    if (threadIdx.x < s) { s1[threadIdx.x] += s1[threadIdx.x + s];
                           s2[threadIdx.x] += s2[threadIdx.x + s]; }
    __syncthreads();
  }
  float mean = s1[0] / HW;
  float var  = s2[0] / HW - mean * mean;
  float rs   = rsqrtf(var + 1e-5f);
  for (int i = threadIdx.x; i < HW; i += 256) {
    float v = (px[i] - mean) * rs;
    if (relu) v = fmaxf(v, 0.f);
    py[i] = v;
  }
}

__global__ void k_batchnorm(const float* __restrict__ x, float* __restrict__ y,
                            int Bn, int C, int HW, int relu) {
  int c = blockIdx.x;
  __shared__ float s1[256], s2[256];
  float a = 0.f, q = 0.f;
  for (int b = 0; b < Bn; ++b) {
    const float* px = x + ((size_t)b * C + c) * HW;
    for (int i = threadIdx.x; i < HW; i += 256) { float v = px[i]; a += v; q += v * v; }
  }
  s1[threadIdx.x] = a; s2[threadIdx.x] = q; __syncthreads();
  for (int s = 128; s > 0; s >>= 1) {
    if (threadIdx.x < s) { s1[threadIdx.x] += s1[threadIdx.x + s];
                           s2[threadIdx.x] += s2[threadIdx.x + s]; }
    __syncthreads();
  }
  float cnt  = (float)Bn * HW;
  float mean = s1[0] / cnt;
  float var  = s2[0] / cnt - mean * mean;
  float rs   = rsqrtf(var + 1e-5f);
  for (int b = 0; b < Bn; ++b) {
    const float* px = x + ((size_t)b * C + c) * HW;
    float*       py = y + ((size_t)b * C + c) * HW;
    for (int i = threadIdx.x; i < HW; i += 256) {
      float v = (px[i] - mean) * rs;
      if (relu) v = fmaxf(v, 0.f);
      py[i] = v;
    }
  }
}

__global__ void k_add_relu(const float* __restrict__ a, const float* __restrict__ b,
                           float* __restrict__ c, long long n) {
  long long i = (long long)blockIdx.x * blockDim.x + threadIdx.x;
  if (i < n) c[i] = fmaxf(a[i] + b[i], 0.f);
}

__global__ void k_split_ctx(const float* __restrict__ ctx, float* __restrict__ h,
                            float* __restrict__ xc, int Bn, int HD, int P) {
  long long i = (long long)blockIdx.x * blockDim.x + threadIdx.x;
  long long tot = (long long)Bn * HD * P;
  if (i < tot) {
    int p = i % P; int c = (i / P) % HD; int b = i / ((long long)HD * P);
    h [((size_t)b * HD + c) * P + p] = tanhf(ctx[((size_t)b * 256 + c)       * P + p]);
    xc[((size_t)b * HD + c) * P + p] = fmaxf(ctx[((size_t)b * 256 + 128 + c) * P + p], 0.f);
  }
}

__global__ void k_init_coords(float* __restrict__ c, int Bn, int hh, int ww) {
  int P = hh * ww;
  long long i = (long long)blockIdx.x * blockDim.x + threadIdx.x;
  if (i < (long long)Bn * P) {
    int p = i % P; int b = i / P;
    c[((size_t)b * 2 + 0) * P + p] = (float)(p % ww);
    c[((size_t)b * 2 + 1) * P + p] = (float)(p / ww);
  }
}

__global__ void k_flow(const float* __restrict__ c, float* __restrict__ f,
                       int Bn, int hh, int ww) {
  int P = hh * ww;
  long long i = (long long)blockIdx.x * blockDim.x + threadIdx.x;
  if (i < (long long)Bn * P) {
    int p = i % P; int b = i / P;
    f[((size_t)b * 2 + 0) * P + p] = c[((size_t)b * 2 + 0) * P + p] - (float)(p % ww);
    f[((size_t)b * 2 + 1) * P + p] = c[((size_t)b * 2 + 1) * P + p] - (float)(p / ww);
  }
}

// 81-tap bilinear lookup in the L2-resident cost volume
__global__ void k_corr_lookup(const float* __restrict__ corr,
                              const float* __restrict__ coords,
                              float* __restrict__ out, int Bn, int hh, int ww) {
  int P = hh * ww;
  long long tot = (long long)Bn * 81 * P;
  long long i = (long long)blockIdx.x * blockDim.x + threadIdx.x;
  if (i >= tot) return;
  int pix = i % P;
  int t   = (i / P) % 81;
  int b   = i / (81LL * P);
  int iy  = pix / ww, jx = pix % ww;
  float cx = coords[((size_t)b * 2 + 0) * P + pix] + (float)(t / 9 - 4);
  float cy = coords[((size_t)b * 2 + 1) * P + pix] + (float)(t % 9 - 4);
  float x0 = floorf(cx), y0 = floorf(cy);
  float tx = cx - x0,   ty = cy - y0;
  const float* base = corr + ((size_t)b * P + (size_t)iy * ww + jx) * P;
  auto g = [&](float fx, float fy) -> float {
    int xi = (int)fx, yi = (int)fy;
    if (xi < 0 || xi > ww - 1 || yi < 0 || yi > hh - 1) return 0.f;
    return base[yi * ww + xi];
  };
  float v = (1.f - tx) * (1.f - ty) * g(x0, y0)
          +        tx  * (1.f - ty) * g(x0 + 1.f, y0)
          + (1.f - tx) *        ty  * g(x0, y0 + 1.f)
          +        tx  *        ty  * g(x0 + 1.f, y0 + 1.f);
  out[((size_t)b * 81 + t) * P + pix] = v;
}

__global__ void k_copy_ch(float* __restrict__ dst, const float* __restrict__ src,
                          int Bn, int Cdst, int coff, int Csrc, int HW) {
  long long tot = (long long)Bn * Csrc * HW;
  long long i = (long long)blockIdx.x * blockDim.x + threadIdx.x;
  if (i < tot) {
    int p = i % HW; int c = (i / HW) % Csrc; int b = i / ((long long)Csrc * HW);
    dst[((size_t)b * Cdst + coff + c) * HW + p] = src[i];
  }
}

__global__ void k_sigmul(const float* __restrict__ r, const float* __restrict__ h,
                         float* __restrict__ rh, long long n) {
  long long i = (long long)blockIdx.x * blockDim.x + threadIdx.x;
  if (i < n) rh[i] = h[i] / (1.f + expf(-r[i]));
}

__global__ void k_gru(float* __restrict__ h, const float* __restrict__ zp,
                      const float* __restrict__ qp, long long n) {
  long long i = (long long)blockIdx.x * blockDim.x + threadIdx.x;
  if (i < n) {
    float z = 1.f / (1.f + expf(-zp[i]));
    h[i] = (1.f - z) * h[i] + z * tanhf(qp[i]);
  }
}

__global__ void k_addto(float* __restrict__ a, const float* __restrict__ b,
                        long long n) {
  long long i = (long long)blockIdx.x * blockDim.x + threadIdx.x;
  if (i < n) a[i] += b[i];
}

// convex 8x upsampling
__global__ void k_up8(const float* __restrict__ mask, const float* __restrict__ flow,
                      float* __restrict__ out, int Bn, int hh, int ww) {
  int H8 = hh * 8, W8 = ww * 8;
  long long tot = (long long)Bn * 2 * H8 * W8;
  long long idx = (long long)blockIdx.x * blockDim.x + threadIdx.x;
  if (idx >= tot) return;
  int x  = idx % W8;
  int y  = (idx / W8) % H8;
  int c  = (idx / ((long long)W8 * H8)) % 2;
  int b  = idx / ((long long)W8 * H8 * 2);
  int i  = y >> 3, r1 = y & 7, j = x >> 3, r2 = x & 7;
  float mv[9], mx = -1e30f;
#pragma unroll
  for (int k = 0; k < 9; ++k) {
    mv[k] = mask[(((size_t)b * 576 + k * 64 + r1 * 8 + r2) * hh + i) * ww + j];
    mx = fmaxf(mx, mv[k]);
  }
  float se = 0.f;
#pragma unroll
  for (int k = 0; k < 9; ++k) { mv[k] = expf(mv[k] - mx); se += mv[k]; }
  float acc = 0.f;
#pragma unroll
  for (int k = 0; k < 9; ++k) {
    int di = k / 3 - 1, dj = k % 3 - 1;
    int ii = i + di, jj = j + dj;
    float f = (ii >= 0 && ii < hh && jj >= 0 && jj < ww)
                  ? 8.f * flow[(((size_t)b * 2 + c) * hh + ii) * ww + jj] : 0.f;
    acc += mv[k] * f;
  }
  out[(((size_t)b * 2 + c) * H8 + y) * W8 + x] = acc / se;
}

// ---------------------------------------------------------------------------
// Host orchestration
// ---------------------------------------------------------------------------
struct Arena { char* base; size_t off; size_t cap; };
static inline void* aalloc(Arena& a, size_t bytes) {
  size_t o = (a.off + 255) & ~(size_t)255;
  a.off = o + bytes;
  return a.base + o;
}
static inline float*  af(Arena& a, size_t n) { return (float*) aalloc(a, n * sizeof(float)); }
static inline half_t* ah(Arena& a, size_t n) { return (half_t*)aalloc(a, n * sizeof(half_t)); }

struct CS { int cin, cout, kh, kw; };
static const CS ENC[17] = {
  {3,64,7,7},{64,64,3,3},{64,64,3,3},{64,64,3,3},{64,64,3,3},
  {64,96,3,3},{96,96,3,3},{64,96,1,1},{96,96,3,3},{96,96,3,3},
  {96,128,3,3},{128,128,3,3},{96,128,1,1},{128,128,3,3},{128,128,3,3},
  {128,160,3,3},{160,256,1,1}};
static const CS UPD[15] = {
  {81,256,1,1},{256,192,3,3},{2,128,7,7},{128,64,3,3},{256,126,3,3},
  {384,128,1,5},{384,128,1,5},{384,128,1,5},{384,128,5,1},{384,128,5,1},{384,128,5,1},
  {128,256,3,3},{256,2,3,3},{128,256,3,3},{256,576,1,1}};

static void conv_launch(hipStream_t s, const float* in, const half_t* wt,
                        const float* bias, float* out, int Bn, int Cin, int H,
                        int W, int Cout, int KH, int KW, int stride, int ph,
                        int pw, int act) {
  int OH = (H + 2 * ph - KH) / stride + 1;
  int OW = (W + 2 * pw - KW) / stride + 1;
  int M = Bn * OH * OW;
  dim3 g((M + BM - 1) / BM, (Cout + BN - 1) / BN);
  k_conv_wmma<<<g, 256, 0, s>>>(in, wt, bias, out, Bn, Cin, H, W, Cout, KH, KW,
                                stride, ph, pw, OH, OW, act);
}

static void norm_launch(hipStream_t s, const float* in, float* out, int Bn,
                        int C, int HW, int normType, int relu) {
  if (normType == 0) k_instnorm<<<Bn * C, 256, 0, s>>>(in, out, HW, relu);
  else               k_batchnorm<<<C, 256, 0, s>>>(in, out, Bn, C, HW, relu);
}

struct EncCtx { const float* const* Bi; half_t* const* WT; int normType; };

static void run_encoder(hipStream_t s, Arena ar, const EncCtx& E,
                        const float* img, int Bn, float* out) {
  size_t cap = (size_t)Bn * 64 * 192 * 256;
  float* bx = af(ar, cap); float* by = af(ar, cap);
  float* bt = af(ar, cap); float* bd = af(ar, cap);
  int H = 384, W = 512, C = 3;
  conv_launch(s, img, E.WT[0], E.Bi[0], bt, Bn, 3, H, W, 64, 7, 7, 2, 3, 3, 0);
  H = 192; W = 256; C = 64;
  norm_launch(s, bt, bx, Bn, 64, H * W, E.normType, 1);

  auto res = [&](int i1, int i2, int idn, int Cout, int stride) {
    int OH = (stride == 2) ? H / 2 : H, OW = (stride == 2) ? W / 2 : W;
    conv_launch(s, bx, E.WT[i1], E.Bi[i1], bt, Bn, C, H, W, Cout, 3, 3, stride, 1, 1, 0);
    norm_launch(s, bt, by, Bn, Cout, OH * OW, E.normType, 1);
    conv_launch(s, by, E.WT[i2], E.Bi[i2], bt, Bn, Cout, OH, OW, Cout, 3, 3, 1, 1, 1, 0);
    norm_launch(s, bt, by, Bn, Cout, OH * OW, E.normType, 1);
    const float* xin = bx;
    if (stride > 1) {
      conv_launch(s, bx, E.WT[idn], E.Bi[idn], bt, Bn, C, H, W, Cout, 1, 1, stride, 0, 0, 0);
      norm_launch(s, bt, bd, Bn, Cout, OH * OW, E.normType, 0);
      xin = bd;
    }
    long long n = (long long)Bn * Cout * OH * OW;
    k_add_relu<<<(unsigned)((n + 255) / 256), 256, 0, s>>>(xin, by, bx, n);
    H = OH; W = OW; C = Cout;
  };
  res(1, 2, -1, 64, 1);   res(3, 4, -1, 64, 1);
  res(5, 6, 7, 96, 2);    res(8, 9, -1, 96, 1);
  res(10, 11, 12, 128, 2); res(13, 14, -1, 128, 1);
  conv_launch(s, bx, E.WT[15], E.Bi[15], bt, Bn, 128, H, W, 160, 3, 3, 1, 1, 1, 0);
  norm_launch(s, bt, by, Bn, 160, H * W, E.normType, 1);
  conv_launch(s, by, E.WT[16], E.Bi[16], out, Bn, 160, H, W, 256, 1, 1, 1, 0, 0, 0);
}

#define GS(n) (unsigned)(((n) + 255) / 256), 256, 0, stream

extern "C" void kernel_launch(void* const* d_in, const int* in_sizes, int n_in,
                              void* d_out, int out_size, void* d_ws, size_t ws_size,
                              hipStream_t stream) {
  (void)in_sizes; (void)n_in; (void)out_size;
  const int Bn = 2, hh = 48, ww = 64, P = hh * ww, D = 256, HD = 128, ITERS = 12;

  const float* img1 = (const float*)d_in[0];
  const float* img2 = (const float*)d_in[1];
  const float* Wp[49]; const float* Bp[49];
  for (int i = 0; i < 49; ++i) {
    Wp[i] = (const float*)d_in[2 + 2 * i];
    Bp[i] = (const float*)d_in[3 + 2 * i];
  }

  Arena ar{(char*)d_ws, 0, ws_size};

  half_t* WT[49];
  for (int i = 0; i < 49; ++i) {
    CS c = (i < 17) ? ENC[i] : (i < 34) ? ENC[i - 17] : UPD[i - 34];
    int K = c.cin * c.kh * c.kw, N = c.cout;
    WT[i] = ah(ar, (size_t)K * N);
    k_w2half<<<GS((long long)K * N)>>>(Wp[i], WT[i], K, N);
  }

  float*  fmap1  = af(ar, (size_t)Bn * D * P);
  float*  fmap2  = af(ar, (size_t)Bn * D * P);
  float*  ctx    = af(ar, (size_t)Bn * D * P);
  float*  hstate = af(ar, (size_t)Bn * HD * P);
  float*  xctx   = af(ar, (size_t)Bn * HD * P);
  float*  corr   = af(ar, (size_t)Bn * P * P);      // 75 MB: L2-resident
  half_t* a16    = ah(ar, (size_t)Bn * P * D);
  half_t* b16    = ah(ar, (size_t)Bn * D * P);
  float*  coords = af(ar, (size_t)Bn * 2 * P);
  float*  cfeat  = af(ar, (size_t)Bn * 81 * P);
  float*  flow   = af(ar, (size_t)Bn * 2 * P);
  float*  cor1   = af(ar, (size_t)Bn * 256 * P);
  float*  cor    = af(ar, (size_t)Bn * 192 * P);
  float*  flo1   = af(ar, (size_t)Bn * 128 * P);
  float*  flo    = af(ar, (size_t)Bn * 64 * P);
  float*  cf     = af(ar, (size_t)Bn * 256 * P);
  float*  mfeat  = af(ar, (size_t)Bn * 126 * P);
  float*  xfeat  = af(ar, (size_t)Bn * 256 * P);
  float*  hx     = af(ar, (size_t)Bn * 384 * P);
  float*  zp     = af(ar, (size_t)Bn * HD * P);
  float*  rp     = af(ar, (size_t)Bn * HD * P);
  float*  qp     = af(ar, (size_t)Bn * HD * P);
  float*  rh     = af(ar, (size_t)Bn * HD * P);
  float*  rhx    = af(ar, (size_t)Bn * 384 * P);
  float*  t256   = af(ar, (size_t)Bn * 256 * P);
  float*  dflow  = af(ar, (size_t)Bn * 2 * P);
  float*  maskb  = af(ar, (size_t)Bn * 576 * P);

  EncCtx Ef{Bp + 0,  WT + 0,  0};   // fnet: instance norm
  EncCtx Ec{Bp + 17, WT + 17, 1};   // cnet: batch norm
  run_encoder(stream, ar, Ef, img1, Bn, fmap1);
  run_encoder(stream, ar, Ef, img2, Bn, fmap2);
  run_encoder(stream, ar, Ec, img1, Bn, ctx);

  k_split_ctx<<<GS((long long)Bn * HD * P)>>>(ctx, hstate, xctx, Bn, HD, P);

  // correlation volume: per-batch 3072x3072x256 WMMA GEMM, alpha = 1/sqrt(256)
  k_t_f2h<<<GS((long long)Bn * D * P)>>>(fmap1, a16, Bn, D, P);
  k_f2h <<<GS((long long)Bn * D * P)>>>(fmap2, b16, (long long)Bn * D * P);
  for (int b = 0; b < Bn; ++b) {
    dim3 g((P + BM - 1) / BM, (P + BNG - 1) / BNG);
    k_gemm_wmma<<<g, 256, 0, stream>>>(a16 + (size_t)b * P * D,
                                       b16 + (size_t)b * D * P,
                                       corr + (size_t)b * P * P,
                                       P, P, D, 0.0625f);
  }
  k_init_coords<<<GS((long long)Bn * P)>>>(coords, Bn, hh, ww);

  const long long nH = (long long)Bn * HD * P;
  const float* Ub[15]; half_t* Uw[15];
  for (int i = 0; i < 15; ++i) { Ub[i] = Bp[34 + i]; Uw[i] = WT[34 + i]; }

  for (int it = 0; it < ITERS; ++it) {
    k_corr_lookup<<<GS((long long)Bn * 81 * P)>>>(corr, coords, cfeat, Bn, hh, ww);
    k_flow<<<GS((long long)Bn * P)>>>(coords, flow, Bn, hh, ww);

    conv_launch(stream, cfeat, Uw[0], Ub[0], cor1, Bn, 81, hh, ww, 256, 1, 1, 1, 0, 0, 1);
    conv_launch(stream, cor1,  Uw[1], Ub[1], cor,  Bn, 256, hh, ww, 192, 3, 3, 1, 1, 1, 1);
    conv_launch(stream, flow,  Uw[2], Ub[2], flo1, Bn, 2,   hh, ww, 128, 7, 7, 1, 3, 3, 1);
    conv_launch(stream, flo1,  Uw[3], Ub[3], flo,  Bn, 128, hh, ww, 64,  3, 3, 1, 1, 1, 1);
    k_copy_ch<<<GS((long long)Bn * 192 * P)>>>(cf, cor, Bn, 256, 0,   192, P);
    k_copy_ch<<<GS((long long)Bn * 64  * P)>>>(cf, flo, Bn, 256, 192, 64,  P);
    conv_launch(stream, cf, Uw[4], Ub[4], mfeat, Bn, 256, hh, ww, 126, 3, 3, 1, 1, 1, 1);
    k_copy_ch<<<GS((long long)Bn * 128 * P)>>>(xfeat, xctx,  Bn, 256, 0,   128, P);
    k_copy_ch<<<GS((long long)Bn * 126 * P)>>>(xfeat, mfeat, Bn, 256, 128, 126, P);
    k_copy_ch<<<GS((long long)Bn * 2   * P)>>>(xfeat, flow,  Bn, 256, 254, 2,   P);

    for (int pass = 0; pass < 2; ++pass) {
      int kz = 5 + 3 * pass;   // convz1/r1/q1 then convz2/r2/q2
      int kh2 = (pass == 0) ? 1 : 5, kw2 = (pass == 0) ? 5 : 1;
      int ph2 = (pass == 0) ? 0 : 2, pw2 = (pass == 0) ? 2 : 0;
      k_copy_ch<<<GS((long long)Bn * 128 * P)>>>(hx, hstate, Bn, 384, 0,   128, P);
      k_copy_ch<<<GS((long long)Bn * 256 * P)>>>(hx, xfeat,  Bn, 384, 128, 256, P);
      conv_launch(stream, hx, Uw[kz],     Ub[kz],     zp, Bn, 384, hh, ww, 128, kh2, kw2, 1, ph2, pw2, 0);
      conv_launch(stream, hx, Uw[kz + 1], Ub[kz + 1], rp, Bn, 384, hh, ww, 128, kh2, kw2, 1, ph2, pw2, 0);
      k_sigmul<<<GS(nH)>>>(rp, hstate, rh, nH);
      k_copy_ch<<<GS((long long)Bn * 128 * P)>>>(rhx, rh,    Bn, 384, 0,   128, P);
      k_copy_ch<<<GS((long long)Bn * 256 * P)>>>(rhx, xfeat, Bn, 384, 128, 256, P);
      conv_launch(stream, rhx, Uw[kz + 2], Ub[kz + 2], qp, Bn, 384, hh, ww, 128, kh2, kw2, 1, ph2, pw2, 0);
      k_gru<<<GS(nH)>>>(hstate, zp, qp, nH);
    }

    conv_launch(stream, hstate, Uw[11], Ub[11], t256,  Bn, 128, hh, ww, 256, 3, 3, 1, 1, 1, 1);
    conv_launch(stream, t256,   Uw[12], Ub[12], dflow, Bn, 256, hh, ww, 2,   3, 3, 1, 1, 1, 0);
    k_addto<<<GS((long long)Bn * 2 * P)>>>(coords, dflow, (long long)Bn * 2 * P);
  }

  k_flow<<<GS((long long)Bn * P)>>>(coords, flow, Bn, hh, ww);
  conv_launch(stream, hstate, Uw[13], Ub[13], t256,  Bn, 128, hh, ww, 256, 3, 3, 1, 1, 1, 1);
  conv_launch(stream, t256,   Uw[14], Ub[14], maskb, Bn, 256, hh, ww, 576, 1, 1, 1, 0, 0, 0);
  k_up8<<<GS((long long)Bn * 2 * hh * 8 * ww * 8)>>>(maskb, flow, (float*)d_out, Bn, hh, ww);
}
